// MHSelfAttention_39676907888250
// MI455X (gfx1250) — matrix operand order repaired and checked by reading, not compiled, not run on hardware
//
#include <hip/hip_runtime.h>
#include <hip/hip_bf16.h>

typedef __attribute__((ext_vector_type(8)))  float          v8f;
typedef __attribute__((ext_vector_type(16))) __bf16         v16bf;
typedef __attribute__((ext_vector_type(4)))  float          v4f;
typedef __attribute__((ext_vector_type(4)))  unsigned short u16x4;
typedef __attribute__((ext_vector_type(8)))  unsigned short u16x8;
typedef __attribute__((ext_vector_type(16))) unsigned short u16x16;

#define BATCH 2
#define SEQ   2048
#define EMB   1024
#define NH    16
#define HD    64
#define MROWS (BATCH * SEQ)   // 4096

// ---------------------------------------------------------------------------
// helpers
// ---------------------------------------------------------------------------
__device__ __forceinline__ unsigned short f2bf(float f) {
  unsigned int u = __builtin_bit_cast(unsigned int, f);
  u += 0x7FFFu + ((u >> 16) & 1u);               // round-to-nearest-even
  return (unsigned short)(u >> 16);
}

__device__ __forceinline__ v16bf bfvec(u16x8 lo, u16x8 hi) {
  u16x16 w;
#pragma unroll
  for (int i = 0; i < 8; ++i) { w[i] = lo[i]; w[i + 8] = hi[i]; }
  return __builtin_bit_cast(v16bf, w);
}

// Low 32 bits of a flat pointer into LDS are the LDS byte offset (ISA 10.2).
__device__ __forceinline__ unsigned lds_addr32(const void* p) {
  return (unsigned)(unsigned long long)(uintptr_t)p;
}

// Async DMA: 16 bytes global -> LDS per lane, tracked by ASYNCcnt.
__device__ __forceinline__ void async_copy_b128(unsigned lds,
                                                const unsigned short* g) {
  asm volatile("global_load_async_to_lds_b128 %0, %1, off"
               :: "v"(lds), "v"((unsigned long long)(uintptr_t)g)
               : "memory");
}

// ---------------------------------------------------------------------------
// fp32 -> bf16 conversion (vectorized 4-wide)
// ---------------------------------------------------------------------------
__global__ void cvt_bf16_kernel(const float* __restrict__ src,
                                unsigned short* __restrict__ dst, int n) {
  int i = (blockIdx.x * blockDim.x + threadIdx.x) * 4;
  if (i + 3 < n) {
    v4f f = *(const v4f*)(src + i);
    u16x4 o;
#pragma unroll
    for (int j = 0; j < 4; ++j) o[j] = f2bf(f[j]);
    *(u16x4*)(dst + i) = o;
  } else {
    for (int j = i; j < n; ++j) dst[j] = f2bf(src[j]);
  }
}

// ---------------------------------------------------------------------------
// GEMM: out[M,N] = (A[M,K] * W[N,K]^T + bias[N]) * scale
// One wave computes a 16x64 output strip: one A fragment reused over 4 B
// tiles, 4 WMMAs per 32-deep k-step.
// MODE 0: out bf16, scatter to [B,H,S,D]   (Q, K projections)
// MODE 1: out bf16, scatter to [B,H,D,S]   (V projection, pre-transposed)
// MODE 2: out fp32, row-major [M,N]        (final projection -> d_out)
// ---------------------------------------------------------------------------
template <int MODE>
__global__ __launch_bounds__(256) void gemm_bf16_kernel(
    const unsigned short* __restrict__ A, const unsigned short* __restrict__ W,
    const float* __restrict__ bias, void* __restrict__ outp, float scale) {
  const int lane  = threadIdx.x & 31;
  const int wave  = blockIdx.x * (blockDim.x >> 5) + (threadIdx.x >> 5);
  const int strips = EMB / 64;                 // 16 strips of 64 cols
  const int tileM = wave / strips;             // 0..255
  const int strip = wave % strips;
  const int lo16  = lane & 15;
  const int hi    = lane >> 4;

  // A fragment: lane = row, two contiguous 8-elem K runs (k0+8*hi, +16)
  const unsigned short* arow = A + (size_t)(tileM * 16 + lo16) * EMB + hi * 8;
  // B fragments: lane = output col N -> row of W; two 8-elem K runs (k0+8*hi, +16)
  const unsigned short* brow[4];
#pragma unroll
  for (int c = 0; c < 4; ++c)
    brow[c] = W + (size_t)(strip * 64 + c * 16 + lo16) * EMB + hi * 8;

  v8f acc[4] = {{}, {}, {}, {}};
#pragma unroll 2
  for (int k0 = 0; k0 < EMB; k0 += 32) {
    u16x8 alo = *(const u16x8*)(arow + k0);
    u16x8 ahi = *(const u16x8*)(arow + k0 + 16);
    v16bf af  = bfvec(alo, ahi);
    __builtin_prefetch(arow + k0 + 128, 0, 3);
#pragma unroll
    for (int c = 0; c < 4; ++c) {
      u16x8 blo = *(const u16x8*)(brow[c] + k0);
      u16x8 bhi = *(const u16x8*)(brow[c] + k0 + 16);
      acc[c] = __builtin_amdgcn_wmma_f32_16x16x32_bf16(
          false, af, false, bfvec(blo, bhi), (short)0, acc[c], false, false);
    }
  }

#pragma unroll
  for (int c = 0; c < 4; ++c) {
    const int   col = strip * 64 + c * 16 + lo16;
    const float bv  = bias[col];
#pragma unroll
    for (int i = 0; i < 8; ++i) {
      const int   row = tileM * 16 + i + hi * 8;  // C layout: vgpr i, half
      const float v   = (acc[c][i] + bv) * scale;
      if (MODE == 2) {
        ((float*)outp)[(size_t)row * EMB + col] = v;
      } else {
        const int b = row >> 11;        // / SEQ
        const int s = row & (SEQ - 1);
        const int h = col >> 6;         // / HD
        const int d = col & (HD - 1);
        size_t idx;
        if (MODE == 0) idx = ((((size_t)b * NH + h) * SEQ) + s) * HD + d;
        else           idx = ((((size_t)b * NH + h) * HD) + d) * SEQ + s;
        ((unsigned short*)outp)[idx] = f2bf(v);
      }
    }
  }
}

// ---------------------------------------------------------------------------
// Flash attention, block-cooperative version.
// A block's 8 waves share one (b,h) head and own 8 consecutive 16-row query
// tiles.  64-wide key tiles (K: [t][d], Vt: [d][t]) are staged into LDS once
// per block with async global->LDS DMA (ASYNCcnt), double-buffered so the
// copy of tile tt+1 overlaps the WMMA/softmax of tile tt.
// Scale 1/sqrt(D) is pre-folded into Q. 16 WMMAs per key tile per wave.
// ---------------------------------------------------------------------------
__global__ __launch_bounds__(256) void flash_attn_kernel(
    const unsigned short* __restrict__ Qb, const unsigned short* __restrict__ Kb,
    const unsigned short* __restrict__ Vtb, unsigned short* __restrict__ Yb) {
  __shared__ __align__(16) unsigned short kbuf[2][64 * 64];  // 16 KB
  __shared__ __align__(16) unsigned short vbuf[2][64 * 64];  // 16 KB
  __shared__ __align__(16) unsigned short plds[8][16 * 64];  // 16 KB

  const int lane   = threadIdx.x & 31;
  const int wslot  = threadIdx.x >> 5;
  const int bh     = blockIdx.x >> 4;        // head index (0..31)
  const int sgroup = blockIdx.x & 15;        // group of 8 query tiles
  const int stile  = sgroup * 8 + wslot;     // this wave's 16-row query tile
  const int h      = bh % NH;
  const int b      = bh / NH;
  const int lo16   = lane & 15;
  const int hi     = lane >> 4;

  const size_t headQK = ((size_t)b * NH + h) * SEQ * HD;
  const size_t headVt = ((size_t)b * NH + h) * HD * SEQ;
  const unsigned short* Kh = Kb + headQK;
  const unsigned short* Vh = Vtb + headVt;

  // Q A-fragments for k-slices [0,32) and [32,64)  (already scaled by 1/8)
  const unsigned short* qrow =
      Qb + headQK + (size_t)(stile * 16 + lo16) * HD + hi * 8;
  v16bf qf[2];
#pragma unroll
  for (int kk = 0; kk < 2; ++kk) {
    u16x8 lo = *(const u16x8*)(qrow + kk * 32);
    u16x8 hx = *(const u16x8*)(qrow + kk * 32 + 16);
    qf[kk] = bfvec(lo, hx);
  }

  v8f   acc[4] = {{}, {}, {}, {}};
  float m[8], l[8];
#pragma unroll
  for (int i = 0; i < 8; ++i) { m[i] = -1e30f; l[i] = 0.f; }

  const int ntt_w   = ((stile * 16 + 15) >> 6) + 1;  // this wave's key tiles
  const int ntt_blk = 2 * sgroup + 2;                // block max key tiles

  // cooperative async staging of one 64-key tile (512 x 16B per array)
  auto stage = [&](int buf, int tbase) {
#pragma unroll
    for (int r = 0; r < 2; ++r) {
      const int chunk = threadIdx.x + r * 256;  // 0..511
      const int row   = chunk >> 3;             // 0..63
      const int cc    = (chunk & 7) * 8;        // element column
      async_copy_b128(lds_addr32(&kbuf[buf][row * 64 + cc]),
                      Kh + (size_t)(tbase + row) * HD + cc);
      async_copy_b128(lds_addr32(&vbuf[buf][row * 64 + cc]),
                      Vh + (size_t)row * SEQ + tbase + cc);
    }
  };

  stage(0, 0);  // prologue: tile 0 in flight

  for (int tt = 0; tt < ntt_blk; ++tt) {
    const int tbase = tt * 64;
    // issue next tile's DMA, then wait for current tile (4 async ops/wave
    // per tile; async loads complete in order)
    if (tt + 1 < ntt_blk) {
      stage((tt + 1) & 1, tbase + 64);
      asm volatile("s_wait_asynccnt 4" ::: "memory");
    } else {
      asm volatile("s_wait_asynccnt 0" ::: "memory");
    }
    __syncthreads();  // current tile visible to all waves

    if (tt < ntt_w) {
      const unsigned short* kb = kbuf[tt & 1];
      const unsigned short* vb = vbuf[tt & 1];

      // ---- scores S = Q * K^T (16x64 tile = four 16x16 WMMA outputs) ----
      v8f s[4] = {{}, {}, {}, {}};
#pragma unroll
      for (int kk = 0; kk < 2; ++kk) {
#pragma unroll
        for (int nt = 0; nt < 4; ++nt) {
          const unsigned short* kr =
              kb + (nt * 16 + lo16) * 64 + kk * 32 + hi * 8;
          u16x8 bl  = *(const u16x8*)(kr);
          u16x8 bhx = *(const u16x8*)(kr + 16);
          s[nt] = __builtin_amdgcn_wmma_f32_16x16x32_bf16(
              false, qf[kk], false, bfvec(bl, bhx), (short)0, s[nt], false,
              false);
        }
      }

      // ---- causal mask + online softmax (row reductions, 16-lane halves) --
#pragma unroll
      for (int i = 0; i < 8; ++i) {
        const int row = stile * 16 + i + hi * 8;
        const int c0  = tbase + lo16;
        float a0 = (c0      <= row) ? s[0][i] : -1e30f;
        float a1 = (c0 + 16 <= row) ? s[1][i] : -1e30f;
        float a2 = (c0 + 32 <= row) ? s[2][i] : -1e30f;
        float a3 = (c0 + 48 <= row) ? s[3][i] : -1e30f;
        float mt = fmaxf(fmaxf(a0, a1), fmaxf(a2, a3));
        mt = fmaxf(mt, __shfl_xor(mt, 1));
        mt = fmaxf(mt, __shfl_xor(mt, 2));
        mt = fmaxf(mt, __shfl_xor(mt, 4));
        mt = fmaxf(mt, __shfl_xor(mt, 8));
        const float mnew = fmaxf(m[i], mt);
        const float corr = __expf(m[i] - mnew);
        const float p0   = __expf(a0 - mnew);
        const float p1   = __expf(a1 - mnew);
        const float p2   = __expf(a2 - mnew);
        const float p3   = __expf(a3 - mnew);
        float rs = (p0 + p1) + (p2 + p3);
        rs += __shfl_xor(rs, 1);
        rs += __shfl_xor(rs, 2);
        rs += __shfl_xor(rs, 4);
        rs += __shfl_xor(rs, 8);
        l[i] = l[i] * corr + rs;
        m[i] = mnew;
#pragma unroll
        for (int nc = 0; nc < 4; ++nc) acc[nc][i] *= corr;
        const int pbase = (i + hi * 8) * 64 + lo16;
        plds[wslot][pbase]      = f2bf(p0);
        plds[wslot][pbase + 16] = f2bf(p1);
        plds[wslot][pbase + 32] = f2bf(p2);
        plds[wslot][pbase + 48] = f2bf(p3);
      }

      // ---- reshape P (C layout -> A fragment) via LDS, then O += P * V ----
#pragma unroll
      for (int kk2 = 0; kk2 < 2; ++kk2) {     // two 32-deep k-steps over t
        const unsigned short* pr = &plds[wslot][lo16 * 64 + kk2 * 32 + hi * 8];
        u16x8 plo = *(const u16x8*)(pr);
        u16x8 phi = *(const u16x8*)(pr + 16);
        v16bf pf  = bfvec(plo, phi);
#pragma unroll
        for (int nc = 0; nc < 4; ++nc) {
          const unsigned short* vr =
              vb + (nc * 16 + lo16) * 64 + kk2 * 32 + hi * 8;
          u16x8 vlo = *(const u16x8*)(vr);
          u16x8 vhi = *(const u16x8*)(vr + 16);
          acc[nc] = __builtin_amdgcn_wmma_f32_16x16x32_bf16(
              false, pf, false, bfvec(vlo, vhi), (short)0, acc[nc], false,
              false);
        }
      }
    }

    __syncthreads();  // all waves done reading before buffer is overwritten
  }

  // ---- epilogue: O / l -> Y[B,S,E] bf16 ----
#pragma unroll
  for (int i = 0; i < 8; ++i) {
    const int    s    = stile * 16 + i + hi * 8;
    const float  inv  = 1.f / l[i];
    const size_t base = ((size_t)b * SEQ + s) * EMB + h * HD + lo16;
#pragma unroll
    for (int nc = 0; nc < 4; ++nc)
      Yb[base + nc * 16] = f2bf(acc[nc][i] * inv);
  }
}

// ---------------------------------------------------------------------------
// launch
// ---------------------------------------------------------------------------
extern "C" void kernel_launch(void* const* d_in, const int* in_sizes, int n_in,
                              void* d_out, int out_size, void* d_ws,
                              size_t ws_size, hipStream_t stream) {
  (void)in_sizes; (void)n_in; (void)out_size; (void)ws_size;
  const float* x  = (const float*)d_in[0];
  const float* Wq = (const float*)d_in[1];
  const float* bq = (const float*)d_in[2];
  const float* Wk = (const float*)d_in[3];
  const float* bk = (const float*)d_in[4];
  const float* Wv = (const float*)d_in[5];
  const float* bv = (const float*)d_in[6];
  const float* Wp = (const float*)d_in[7];
  const float* bp = (const float*)d_in[8];
  float* out = (float*)d_out;

  unsigned short* ws  = (unsigned short*)d_ws;
  unsigned short* xb  = ws;                                   // [MROWS,EMB]
  unsigned short* Wqb = xb  + (size_t)MROWS * EMB;
  unsigned short* Wkb = Wqb + (size_t)EMB * EMB;
  unsigned short* Wvb = Wkb + (size_t)EMB * EMB;
  unsigned short* Wpb = Wvb + (size_t)EMB * EMB;
  unsigned short* Qb  = Wpb + (size_t)EMB * EMB;              // [B,H,S,D]
  unsigned short* Kb  = Qb  + (size_t)MROWS * EMB;            // [B,H,S,D]
  unsigned short* Vtb = Kb  + (size_t)MROWS * EMB;            // [B,H,D,S]
  unsigned short* Yb  = Vtb + (size_t)MROWS * EMB;            // [B,S,E]

  auto cvt = [&](const float* s, unsigned short* d, int n) {
    const int threads = 256;
    const int blocks  = (n + threads * 4 - 1) / (threads * 4);
    cvt_bf16_kernel<<<blocks, threads, 0, stream>>>(s, d, n);
  };
  cvt(x,  xb,  MROWS * EMB);
  cvt(Wq, Wqb, EMB * EMB);
  cvt(Wk, Wkb, EMB * EMB);
  cvt(Wv, Wvb, EMB * EMB);
  cvt(Wp, Wpb, EMB * EMB);

  const int gwaves  = (MROWS / 16) * (EMB / 64);  // 4096 waves (16x64 strips)
  const int gblocks = gwaves / 8;                 // 8 waves per 256-thr block
  const float qscale = 0.125f;                    // 1/sqrt(HD)
  gemm_bf16_kernel<0><<<gblocks, 256, 0, stream>>>(xb, Wqb, bq, (void*)Qb, qscale);
  gemm_bf16_kernel<0><<<gblocks, 256, 0, stream>>>(xb, Wkb, bk, (void*)Kb, 1.0f);
  gemm_bf16_kernel<1><<<gblocks, 256, 0, stream>>>(xb, Wvb, bv, (void*)Vtb, 1.0f);

  // 512 blocks: one (b,h) head + 8 consecutive query tiles per block
  flash_attn_kernel<<<BATCH * NH * 16, 256, 0, stream>>>(Qb, Kb, Vtb, Yb);

  gemm_bf16_kernel<2><<<gblocks, 256, 0, stream>>>(Yb, Wpb, bp, (void*)out, 1.0f);
}
